// PerformerTransact_45500883534221
// MI455X (gfx1250) — matrix operand order repaired and checked by reading, not compiled, not run on hardware
//
#include <hip/hip_runtime.h>
#include <math.h>

// ---------------- problem constants ----------------
#define B_     2048
#define N_     15
#define DIM_   512
#define H_     8
#define GH_    4
#define DH_    64
#define M_     256
#define L_     12
#define VOCAB_ 1024
#define FFIN_  4096
#define FFHID_ 2048
#define T_     (B_*N_)          // 30720 tokens

// ---------------- WMMA types ----------------
typedef __attribute__((ext_vector_type(16))) __bf16 v16bf;
typedef __attribute__((ext_vector_type(8)))  float  v8f;

union Frag16 { v16bf v; unsigned int u[8]; };

// f32 -> packed bf16 pair; plain casts let the backend emit the
// hardware packed conversion (v_cvt_pk_bf16_f32) instead of bit-twiddling.
__device__ __forceinline__ unsigned int pack_bf2(float a, float b) {
    union { __bf16 h[2]; unsigned int u; } r;
    r.h[0] = (__bf16)a;
    r.h[1] = (__bf16)b;
    return r.u;
}

// =====================================================================
// Generic GEMM: C[Mr,Nc] = A[Mr,K] * Bw[K,Nc] (+bias[col]) (+resid) (+relu)
// Mr%128==0, Nc%64==0, K%32==0 hold for every call in this model.
// bf16 inputs, f32 accumulate via v_wmma_f32_16x16x32_bf16.
// Block: 256 thr = 8 waves; block tile 128x64; wave tile 32x32 (4 wmma):
// A fragments reused across 2 N-tiles, B fragments across 2 M-tiles
// -> 8 LDS dwords per WMMA.
// =====================================================================
template <bool HasBias, bool HasRes, bool Relu>
__global__ __launch_bounds__(256)
void gemm_bf16_kernel(const float* __restrict__ A, const float* __restrict__ Bw,
                      float* __restrict__ C, int Mr, int Nc, int K,
                      const float* __restrict__ bias,
                      const float* __restrict__ resid)
{
    __shared__ unsigned int As[128 * 17];  // 128 rows x 16 packed K-pairs, pad 17
    __shared__ unsigned int Bs[32 * 33];   // 16 packed K-pair rows x 64 cols, row stride 65

    const int tid  = threadIdx.x;
    const int lane = tid & 31;
    const int wid  = tid >> 5;
    const int wm   = wid >> 1;     // 0..3  -> 32-row strip
    const int wn   = wid & 1;      // 0..1  -> 32-col strip
    const int bm   = blockIdx.y;
    const int bn   = blockIdx.x;

    v8f c00 = {}, c01 = {}, c10 = {}, c11 = {};

    for (int kk = 0; kk < K; kk += 32) {
        // stage A tile (128x32 f32 -> packed bf16 pairs), coalesced b64 loads
        #pragma unroll
        for (int i = 0; i < 8; ++i) {
            int pi = tid + i * 256;              // 0..2047 pair index
            int r = pi >> 4, p = pi & 15;
            const float* ap = A + (size_t)(bm * 128 + r) * K + (kk + 2 * p);
            As[r * 17 + p] = pack_bf2(ap[0], ap[1]);
        }
        // stage B tile (32x64) as packed K-pairs per column: Bs[kp*65 + n]
        #pragma unroll
        for (int i = 0; i < 4; ++i) {
            int pi = tid + i * 256;
            int kp = pi >> 6, n = pi & 63;
            const float* bp = Bw + (size_t)(kk + 2 * kp) * Nc + (bn * 64 + n);
            Bs[kp * 65 + n] = pack_bf2(bp[0], bp[Nc]);
        }
        // prefetch next K-step tiles into cache while this step computes
        if (kk + 32 < K) {
            #pragma unroll
            for (int i = 0; i < 8; ++i) {
                int pi = tid + i * 256;
                int r = pi >> 4, p = pi & 15;
                __builtin_prefetch(A + (size_t)(bm * 128 + r) * K + (kk + 32 + 2 * p), 0, 3);
            }
            #pragma unroll
            for (int i = 0; i < 4; ++i) {
                int pi = tid + i * 256;
                int r = pi >> 5, p = pi & 31;
                __builtin_prefetch(Bw + (size_t)(kk + 32 + r) * Nc + (bn * 64 + 2 * p), 0, 3);
            }
        }
        __syncthreads();

        // A fragments: lane m=(lane&15), K groups {0-7,16-23} / {8-15,24-31}
        Frag16 a0, a1, b0, b1;
        const int hf = lane >> 4;
        const int m0 = wm * 32 + (lane & 15);
        #pragma unroll
        for (int j = 0; j < 8; ++j) {
            int p = (j & 3) + hf * 4 + ((j >> 2) << 3);
            a0.u[j] = As[m0 * 17 + p];
            a1.u[j] = As[(m0 + 16) * 17 + p];
        }
        // B fragments: lane n=(lane&15), K groups {0-7,16-23} / {8-15,24-31}
        const int nb = wn * 32 + (lane & 15);      // column of this lane in the wave's 32-col strip
        #pragma unroll
        for (int j = 0; j < 8; ++j) {
            int p = (j & 3) + hf * 4 + ((j >> 2) << 3);
            b0.u[j] = Bs[p * 65 + nb];
            b1.u[j] = Bs[p * 65 + nb + 16];
        }

        c00 = __builtin_amdgcn_wmma_f32_16x16x32_bf16(false, a0.v, false, b0.v,
                                                      (short)0, c00, false, false);
        c01 = __builtin_amdgcn_wmma_f32_16x16x32_bf16(false, a0.v, false, b1.v,
                                                      (short)0, c01, false, false);
        c10 = __builtin_amdgcn_wmma_f32_16x16x32_bf16(false, a1.v, false, b0.v,
                                                      (short)0, c10, false, false);
        c11 = __builtin_amdgcn_wmma_f32_16x16x32_bf16(false, a1.v, false, b1.v,
                                                      (short)0, c11, false, false);
        __syncthreads();
    }

    // C layout: lane -> N, vgpr r -> M (lanes 16-31 are M+8)
    const int hf   = lane >> 4;
    const int row0 = bm * 128 + wm * 32 + hf * 8;
    const int col0 = bn * 64 + wn * 32 + (lane & 15);
    float bia0 = 0.f, bia1 = 0.f;
    if (HasBias) { bia0 = bias[col0]; bia1 = bias[col0 + 16]; }
    #pragma unroll
    for (int mt = 0; mt < 2; ++mt) {
        v8f cn0 = (mt == 0) ? c00 : c10;
        v8f cn1 = (mt == 0) ? c01 : c11;
        #pragma unroll
        for (int r = 0; r < 8; ++r) {
            int row = row0 + mt * 16 + r;
            size_t base = (size_t)row * Nc;
            float v0 = cn0[r], v1 = cn1[r];
            if (HasBias) { v0 += bia0; v1 += bia1; }
            if (HasRes)  { v0 += resid[base + col0]; v1 += resid[base + col0 + 16]; }
            if (Relu)    { v0 = v0 > 0.f ? v0 : 0.f; v1 = v1 > 0.f ? v1 : 0.f; }
            C[base + col0]      = v0;
            C[base + col0 + 16] = v1;
        }
    }
}

// =====================================================================
// Embedding: x = token_emb[idx] + fixed_pos_emb(n, 512); writes x1 and x2
// =====================================================================
__global__ __launch_bounds__(256)
void embed_kernel(const int* __restrict__ idx, const float* __restrict__ emb,
                  float* __restrict__ x1, float* __restrict__ x2)
{
    int i = blockIdx.x * 256 + threadIdx.x;      // over T_*DIM_
    int row = i >> 9, d = i & 511;
    int n = row % N_;
    int tok = idx[row];
    float v = emb[(size_t)tok * DIM_ + d];
    float p;
    if (d < 256) p = sinf((float)n * powf(10000.f, -((float)(2 * d)) / 512.f));
    else         p = cosf((float)n * powf(10000.f, -((float)(2 * (d - 256))) / 512.f));
    v += p;
    x1[i] = v; x2[i] = v;
}

// =====================================================================
// LayerNorm (+optional add) (+optional pre_shift).  One block per batch b.
// pre_shift: out[n,0:170]=ln[n+1], [170:340]=ln[n], [340:510]=ln[n-1], rest ln[n]
// =====================================================================
__global__ __launch_bounds__(256)
void ln_shift_kernel(const float* __restrict__ x, const float* __restrict__ xadd,
                     const float* __restrict__ g, const float* __restrict__ be,
                     float* __restrict__ out, int do_shift)
{
    __shared__ float ln[N_][DIM_];
    const int b = blockIdx.x;
    const int tid = threadIdx.x, lane = tid & 31, wid = tid >> 5;

    for (int i = tid; i < N_ * DIM_; i += 256) {
        int n = i >> 9, d = i & 511;
        size_t ix = ((size_t)(b * N_ + n)) * DIM_ + d;
        float v = x[ix];
        if (xadd) v += xadd[ix];
        ln[n][d] = v;
    }
    __syncthreads();

    for (int n = wid; n < N_; n += 8) {
        float s = 0.f, sq = 0.f;
        for (int d = lane; d < DIM_; d += 32) { float v = ln[n][d]; s += v; sq += v * v; }
        for (int off = 16; off >= 1; off >>= 1) {
            s  += __shfl_xor(s,  off, 32);
            sq += __shfl_xor(sq, off, 32);
        }
        float mu  = s * (1.f / DIM_);
        float var = sq * (1.f / DIM_) - mu * mu;
        float inv = rsqrtf(var + 1e-5f);
        for (int d = lane; d < DIM_; d += 32)
            ln[n][d] = (ln[n][d] - mu) * inv * g[d] + be[d];
    }
    __syncthreads();

    for (int i = tid; i < N_ * DIM_; i += 256) {
        int n = i >> 9, d = i & 511;
        float v;
        if (!do_shift)      v = ln[n][d];
        else if (d < 170)   v = (n + 1 < N_) ? ln[n + 1][d] : 0.f;
        else if (d < 340)   v = ln[n][d];
        else if (d < 510)   v = (n > 0) ? ln[n - 1][d] : 0.f;
        else                v = ln[n][d];
        out[((size_t)(b * N_ + n)) * DIM_ + d] = v;
    }
}

// =====================================================================
// Performer (global) heads, one block per (b,h).  Whole head in LDS.
// Uses (qp kp^T) v associativity since N=15 << M=256.
// =====================================================================
__global__ __launch_bounds__(256)
void global_attn_kernel(const float* __restrict__ q, const float* __restrict__ k,
                        const float* __restrict__ v, const float* __restrict__ proj,
                        float* __restrict__ out)
{
    __shared__ float qs[N_][DH_], ks[N_][DH_], vs[N_][DH_];
    __shared__ float qp[N_][M_], kp[N_][M_];
    __shared__ float S[N_][16];
    __shared__ float diagq[N_], diagk[N_], mxq[N_], mxkr[N_], dinv[N_];
    __shared__ float mxk;

    const int b = blockIdx.x / GH_;
    const int h = blockIdx.x % GH_;
    const int tid = threadIdx.x;

    for (int i = tid; i < N_ * DH_; i += 256) {
        int n = i >> 6, e = i & 63;
        size_t r = ((size_t)(b * N_ + n)) * DIM_ + h * DH_ + e;
        qs[n][e] = q[r]; ks[n][e] = k[r]; vs[n][e] = v[r];
    }
    __syncthreads();

    // rotary (rotate_every_two variant)
    for (int i = tid; i < N_ * (DH_ / 2); i += 256) {
        int n = i >> 5, j = i & 31;
        float f = powf(10000.f, -((float)(2 * j)) / 64.f);
        float ang = (float)n * f, cs = cosf(ang), sn = sinf(ang);
        float q0 = qs[n][2 * j], q1 = qs[n][2 * j + 1];
        qs[n][2 * j]     = q0 * cs - q1 * sn;
        qs[n][2 * j + 1] = q1 * cs + q0 * sn;
        float k0 = ks[n][2 * j], k1 = ks[n][2 * j + 1];
        ks[n][2 * j]     = k0 * cs - k1 * sn;
        ks[n][2 * j + 1] = k1 * cs + k0 * sn;
    }
    __syncthreads();

    const float dn = 0.35355339059327373f;   // 64^-0.25
    for (int i = tid; i < N_ * M_; i += 256) {
        int n = i >> 8, m = i & 255;
        const float* pr = proj + m * DH_;
        float sq = 0.f, sk = 0.f;
        for (int e = 0; e < DH_; ++e) { sq += qs[n][e] * pr[e]; sk += ks[n][e] * pr[e]; }
        qp[n][m] = sq * dn;   // dash_q
        kp[n][m] = sk * dn;   // dash_k
    }
    if (tid < N_) {
        float dq = 0.f, dk = 0.f;
        for (int e = 0; e < DH_; ++e) { dq += qs[tid][e] * qs[tid][e]; dk += ks[tid][e] * ks[tid][e]; }
        diagq[tid] = 0.5f * dn * dn * dq;
        diagk[tid] = 0.5f * dn * dn * dk;
    }
    __syncthreads();

    if (tid < N_) {
        float mq = -1e30f, mk2 = -1e30f;
        for (int m = 0; m < M_; ++m) { mq = fmaxf(mq, qp[tid][m]); mk2 = fmaxf(mk2, kp[tid][m]); }
        mxq[tid] = mq; mxkr[tid] = mk2;
    }
    __syncthreads();
    if (tid == 0) {
        float mk2 = -1e30f;
        for (int n = 0; n < N_; ++n) mk2 = fmaxf(mk2, mxkr[n]);
        mxk = mk2;
    }
    __syncthreads();

    const float ratio = 0.0625f;             // 256^-0.5
    for (int i = tid; i < N_ * M_; i += 256) {
        int n = i >> 8, m = i & 255;
        qp[n][m] = ratio * (expf(qp[n][m] - diagq[n] - mxq[n]) + 1e-4f);
        kp[n][m] = ratio * (expf(kp[n][m] - diagk[n] - mxk)   + 1e-4f);
    }
    __syncthreads();

    for (int i = tid; i < N_ * N_; i += 256) {
        int n = i / N_, j2 = i % N_;
        float s = 0.f;
        for (int m = 0; m < M_; ++m) s += qp[n][m] * kp[j2][m];
        S[n][j2] = s;
    }
    __syncthreads();
    if (tid < N_) {
        float d = 0.f;
        for (int j2 = 0; j2 < N_; ++j2) d += S[tid][j2];
        dinv[tid] = 1.f / d;
    }
    __syncthreads();

    for (int i = tid; i < N_ * DH_; i += 256) {
        int n = i >> 6, e = i & 63;
        float o = 0.f;
        for (int j2 = 0; j2 < N_; ++j2) o += S[n][j2] * vs[j2][e];
        out[((size_t)(b * N_ + n)) * DIM_ + h * DH_ + e] = o * dinv[n];
    }
}

// =====================================================================
// Local softmax heads (rot_half rotary), one block per (b, h-GH)
// =====================================================================
__global__ __launch_bounds__(256)
void local_attn_kernel(const float* __restrict__ q, const float* __restrict__ k,
                       const float* __restrict__ v, float* __restrict__ out)
{
    __shared__ float qs[N_][DH_], ks[N_][DH_], vs[N_][DH_];
    __shared__ float P[N_][16];
    const int b  = blockIdx.x / (H_ - GH_);
    const int h  = GH_ + (blockIdx.x % (H_ - GH_));
    const int tid = threadIdx.x;

    for (int i = tid; i < N_ * DH_; i += 256) {
        int n = i >> 6, e = i & 63;
        size_t r = ((size_t)(b * N_ + n)) * DIM_ + h * DH_ + e;
        qs[n][e] = q[r]; ks[n][e] = k[r]; vs[n][e] = v[r];
    }
    __syncthreads();
    for (int i = tid; i < N_ * (DH_ / 2); i += 256) {
        int n = i >> 5, j = i & 31;
        float f = powf(10000.f, -((float)(2 * j)) / 64.f);
        float ang = (float)n * f, cs = cosf(ang), sn = sinf(ang);
        float qa = qs[n][j], qb = qs[n][j + 32];
        qs[n][j]      = qa * cs - qb * sn;
        qs[n][j + 32] = qb * cs + qa * sn;
        float ka = ks[n][j], kb = ks[n][j + 32];
        ks[n][j]      = ka * cs - kb * sn;
        ks[n][j + 32] = kb * cs + ka * sn;
    }
    __syncthreads();
    for (int i = tid; i < N_ * N_; i += 256) {
        int n = i / N_, j2 = i % N_;
        float s = 0.f;
        for (int e = 0; e < DH_; ++e) s += qs[n][e] * ks[j2][e];
        P[n][j2] = s * 0.125f;                 // 64^-0.5
    }
    __syncthreads();
    if (tid < N_) {
        float mx = -1e30f;
        for (int j2 = 0; j2 < N_; ++j2) mx = fmaxf(mx, P[tid][j2]);
        float ss = 0.f;
        for (int j2 = 0; j2 < N_; ++j2) { float e2 = expf(P[tid][j2] - mx); P[tid][j2] = e2; ss += e2; }
        float inv = 1.f / ss;
        for (int j2 = 0; j2 < N_; ++j2) P[tid][j2] *= inv;
    }
    __syncthreads();
    for (int i = tid; i < N_ * DH_; i += 256) {
        int n = i >> 6, e = i & 63;
        float o = 0.f;
        for (int j2 = 0; j2 < N_; ++j2) o += P[n][j2] * vs[j2][e];
        out[((size_t)(b * N_ + n)) * DIM_ + h * DH_ + e] = o;
    }
}

// GLU: act[r,f] = gelu_exact(h[r,f]) * h[r,2048+f]
__global__ __launch_bounds__(256)
void glu_kernel(const float* __restrict__ h, float* __restrict__ act)
{
    int i = blockIdx.x * 256 + threadIdx.x;      // over T_*FFHID_
    int r = i >> 11, f = i & 2047;
    float a = h[(size_t)r * FFIN_ + f];
    float g = h[(size_t)r * FFIN_ + FFHID_ + f];
    float ge = 0.5f * a * (1.f + erff(a * 0.70710678118654752f));
    act[i] = ge * g;
}

// mean over N of logits -> xa[b, v]
__global__ __launch_bounds__(256)
void mean_kernel(const float* __restrict__ logits, float* __restrict__ xa)
{
    int i = blockIdx.x * 256 + threadIdx.x;      // over B_*VOCAB_
    int b = i >> 10, vc = i & 1023;
    float s = 0.f;
    for (int n = 0; n < N_; ++n) s += logits[((size_t)(b * N_ + n)) * VOCAB_ + vc];
    xa[i] = s * (1.f / N_);
}

// final 64 -> 2 head
__global__ __launch_bounds__(256)
void head_kernel(const float* __restrict__ hs, const float* __restrict__ Wl2,
                 const float* __restrict__ bl2, float* __restrict__ out)
{
    int b = blockIdx.x * 256 + threadIdx.x;
    if (b >= B_) return;
    float o0 = bl2[0], o1 = bl2[1];
    for (int kk = 0; kk < 64; ++kk) {
        float hv = hs[b * 64 + kk];
        o0 += hv * Wl2[kk * 2 + 0];
        o1 += hv * Wl2[kk * 2 + 1];
    }
    out[b * 2 + 0] = o0;
    out[b * 2 + 1] = o1;
}

// =====================================================================
extern "C" void kernel_launch(void* const* d_in, const int* in_sizes, int n_in,
                              void* d_out, int out_size, void* d_ws, size_t ws_size,
                              hipStream_t stream)
{
    (void)in_sizes; (void)n_in; (void)out_size; (void)ws_size;
    const int*   x_mcc = (const int*)  d_in[0];
    const float* temb  = (const float*)d_in[1];
    const float* ln1_g = (const float*)d_in[2];
    const float* ln1_b = (const float*)d_in[3];
    const float* Wq    = (const float*)d_in[4];
    const float* Wk    = (const float*)d_in[5];
    const float* Wv    = (const float*)d_in[6];
    const float* Wo    = (const float*)d_in[7];
    const float* bo    = (const float*)d_in[8];
    const float* proj  = (const float*)d_in[9];
    const float* ln2_g = (const float*)d_in[10];
    const float* ln2_b = (const float*)d_in[11];
    const float* W1    = (const float*)d_in[12];
    const float* b1    = (const float*)d_in[13];
    const float* W2    = (const float*)d_in[14];
    const float* b2    = (const float*)d_in[15];
    const float* lnf_g = (const float*)d_in[16];
    const float* lnf_b = (const float*)d_in[17];
    const float* Wout  = (const float*)d_in[18];
    const float* bout  = (const float*)d_in[19];
    const float* Wl1   = (const float*)d_in[20];
    const float* bl1   = (const float*)d_in[21];
    const float* Wl2   = (const float*)d_in[22];
    const float* bl2   = (const float*)d_in[23];

    float* ws = (float*)d_ws;
    const size_t SLOT = (size_t)T_ * DIM_;
    float* x1  = ws;
    float* x2  = x1 + SLOT;
    float* t   = x2 + SLOT;       // shifted-LN / heads-out; t..kb contiguous
    float* qb  = t  + SLOT;
    float* vb  = qb + SLOT;
    float* kb  = vb + SLOT;
    float* hid = kb + SLOT;                          // [T,4096]
    float* act = t;                                  // aliases t,qb,vb,kb = [T,2048]
    float* logits = hid;                             // [T,1024]
    float* xa  = hid + (size_t)T_ * VOCAB_;          // [2048,1024]
    float* hsm = xa  + (size_t)B_ * VOCAB_;          // [2048,64]

    dim3 blk(256);
    embed_kernel<<<(T_ * DIM_) / 256, blk, 0, stream>>>(x_mcc, temb, x1, x2);

    dim3 g512(DIM_ / 64, T_ / 128);
    dim3 gff(FFIN_ / 64, T_ / 128);

    for (int l = 0; l < L_; ++l) {
        ln_shift_kernel<<<B_, blk, 0, stream>>>(x2, nullptr, ln1_g + l * DIM_, ln1_b + l * DIM_, t, 1);
        gemm_bf16_kernel<false, false, false><<<g512, blk, 0, stream>>>(
            t, Wq + (size_t)l * DIM_ * DIM_, qb, T_, DIM_, DIM_, nullptr, nullptr);
        gemm_bf16_kernel<false, false, false><<<g512, blk, 0, stream>>>(
            t, Wk + (size_t)l * DIM_ * DIM_, kb, T_, DIM_, DIM_, nullptr, nullptr);
        gemm_bf16_kernel<false, false, false><<<g512, blk, 0, stream>>>(
            t, Wv + (size_t)l * DIM_ * DIM_, vb, T_, DIM_, DIM_, nullptr, nullptr);
        global_attn_kernel<<<B_ * GH_, blk, 0, stream>>>(qb, kb, vb, proj + (size_t)l * M_ * DH_, t);
        local_attn_kernel<<<B_ * (H_ - GH_), blk, 0, stream>>>(qb, kb, vb, t);
        // y1 = x1 + heads @ Wo + bo  (bias + residual fused)
        gemm_bf16_kernel<true, true, false><<<g512, blk, 0, stream>>>(
            t, Wo + (size_t)l * DIM_ * DIM_, x1, T_, DIM_, DIM_, bo + l * DIM_, x1);
        ln_shift_kernel<<<B_, blk, 0, stream>>>(x1, nullptr, ln2_g + l * DIM_, ln2_b + l * DIM_, t, 1);
        gemm_bf16_kernel<true, false, false><<<gff, blk, 0, stream>>>(
            t, W1 + (size_t)l * DIM_ * FFIN_, hid, T_, FFIN_, DIM_, b1 + l * FFIN_, nullptr);
        glu_kernel<<<(T_ * FFHID_) / 256, blk, 0, stream>>>(hid, act);
        // y2 = x2 + act @ W2 + b2
        gemm_bf16_kernel<true, true, false><<<g512, blk, 0, stream>>>(
            act, W2 + (size_t)l * FFHID_ * DIM_, x2, T_, DIM_, FFHID_, b2 + l * DIM_, x2);
    }

    // final: LN(x1 + x2) -> t
    ln_shift_kernel<<<B_, blk, 0, stream>>>(x1, x2, lnf_g, lnf_b, t, 0);
    dim3 gout(VOCAB_ / 64, T_ / 128);
    gemm_bf16_kernel<true, false, false><<<gout, blk, 0, stream>>>(
        t, Wout, logits, T_, VOCAB_, DIM_, bout, nullptr);
    mean_kernel<<<(B_ * VOCAB_) / 256, blk, 0, stream>>>(logits, xa);
    dim3 gl1(1, B_ / 128);
    gemm_bf16_kernel<true, false, true><<<gl1, blk, 0, stream>>>(
        xa, Wl1, hsm, B_, 64, VOCAB_, bl1, nullptr);
    head_kernel<<<B_ / 256, blk, 0, stream>>>(hsm, Wl2, bl2, (float*)d_out);
}
